// ConditionalRealNVP_27590869910225
// MI455X (gfx1250) — matrix operand order repaired and checked by reading, not compiled, not run on hardware
//
#include <hip/hip_runtime.h>
#include <hip/hip_bf16.h>
#include <math.h>

typedef __attribute__((ext_vector_type(16))) __bf16 v16bf;
typedef __attribute__((ext_vector_type(8)))  float  v8f;
typedef __attribute__((ext_vector_type(4)))  unsigned int v4u;

#define BROWS 256      // batch rows per workgroup (8 waves x 2 M-tiles of 16)
#define HID   128
#define P1    104      // LDS pitch (halves) for K=96 arrays (208B: 16B aligned, bank-friendly)
#define P2    136      // LDS pitch (halves) for K=128 arrays (272B: 16B aligned, bank-friendly)

// Weight image per net-stage: [W1^T 128*P1 | W2^T 128*P2 | W3^T 16*P2] bf16 halves
#define W2OFF (128 * P1)                        // 13312
#define W3OFF (128 * P1 + 128 * P2)             // 30720
#define IMGH  (128 * P1 + 128 * P2 + 16 * P2)   // 32896 halves used
#define IMGSH 34816                             // padded stride (halves) = 17*4096 B
#define IMGB  (IMGSH * 2)                       // 69632 bytes = 17 uniform 16B chunks/thread

// Load a 32-byte WMMA fragment as two ds_load_b128 from 16B-aligned LDS.
__device__ __forceinline__ v16bf ld_frag(const __bf16* p0, const __bf16* p1) {
  union { struct { v4u lo; v4u hi; } s; v16bf v; } u;
  u.s.lo = *(const v4u*)p0;
  u.s.hi = *(const v4u*)p1;
  return u.v;
}

__device__ __forceinline__ v8f wmma_bf16(v16bf a, v16bf b, v8f c) {
  return __builtin_amdgcn_wmma_f32_16x16x32_bf16(false, a, false, b, (short)0, c,
                                                 false, false);
}

__device__ __forceinline__ float fast_tanh(float x) {
#if __has_builtin(__builtin_amdgcn_tanhf)
  return __builtin_amdgcn_tanhf(x);       // single V_TANH_F32 (TRANS, co-executes with WMMA)
#else
  return tanhf(x);
#endif
}

// tanh-form GELU: ~7 VALU ops incl. 1 TRANS
__device__ __forceinline__ float gelu_fast(float x) {
  float x2 = x * x;
  float u  = 0.7978845608028654f * x * __builtin_fmaf(0.044715f, x2, 1.0f);
  return 0.5f * x * (1.0f + fast_tanh(u));
}

// One wave computes rows [m0,m0+32) x cols [0,128) of act(A*W + b), bf16 out.
// Two M-tiles share every B fragment (halves LDS weight traffic).
template<int KLEN, int PA, int PW>
__device__ __forceinline__ void gemm_tile_act2(const __bf16* a_base,
                                               const __bf16* wt_base,
                                               const float* bias,
                                               __bf16* hout,
                                               int m0, int lane)
{
  const int ml = lane & 15;
  const int hi = lane >> 4;
  v8f acc0[8], acc1[8];
#pragma unroll
  for (int nt = 0; nt < 8; ++nt) {
    acc0[nt] = (v8f){0,0,0,0,0,0,0,0};
    acc1[nt] = (v8f){0,0,0,0,0,0,0,0};
  }

#pragma unroll
  for (int kc = 0; kc < KLEN; kc += 32) {
    const __bf16* ar0 = a_base + (m0 + ml) * PA + kc + hi * 8;
    const __bf16* ar1 = ar0 + 16 * PA;
    v16bf a0 = ld_frag(ar0, ar0 + 16);
    v16bf a1 = ld_frag(ar1, ar1 + 16);
#pragma unroll
    for (int nt = 0; nt < 8; ++nt) {
      const __bf16* wr = wt_base + (nt * 16 + ml) * PW + kc + hi * 8;
      v16bf b = ld_frag(wr, wr + 16);
      acc0[nt] = wmma_bf16(a0, b, acc0[nt]);
      acc1[nt] = wmma_bf16(a1, b, acc1[nt]);
    }
  }

  const int hi8 = hi * 8;
#pragma unroll
  for (int nt = 0; nt < 8; ++nt) {
    const int col = nt * 16 + ml;
    const float bv = bias[col];
#pragma unroll
    for (int r = 0; r < 8; ++r) {
      hout[(m0 + r + hi8) * P2 + col]      = (__bf16)gelu_fast(acc0[nt][r] + bv);
      hout[(m0 + 16 + r + hi8) * P2 + col] = (__bf16)gelu_fast(acc1[nt][r] + bv);
    }
  }
}

// Head GEMM: rows [m0,m0+32) x W3t[16][128] (cols 0,1 valid) -> dst[row][2] f32.
__device__ __forceinline__ void gemm_head2(const __bf16* a_base,
                                           const __bf16* w3t,
                                           const float* b3,
                                           float* dst, int m0, int lane)
{
  const int ml = lane & 15;
  const int hi = lane >> 4;
  v8f acc0 = (v8f){0,0,0,0,0,0,0,0};
  v8f acc1 = (v8f){0,0,0,0,0,0,0,0};
#pragma unroll
  for (int kc = 0; kc < HID; kc += 32) {
    const __bf16* ar0 = a_base + (m0 + ml) * P2 + kc + hi * 8;
    const __bf16* ar1 = ar0 + 16 * P2;
    v16bf a0 = ld_frag(ar0, ar0 + 16);
    v16bf a1 = ld_frag(ar1, ar1 + 16);
    const __bf16* wr = w3t + ml * P2 + kc + hi * 8;
    v16bf b = ld_frag(wr, wr + 16);
    acc0 = wmma_bf16(a0, b, acc0);
    acc1 = wmma_bf16(a1, b, acc1);
  }
  if (ml < 2) {
    const float bv = b3[ml];
#pragma unroll
    for (int r = 0; r < 8; ++r) {
      dst[(m0 + r + hi * 8) * 2 + ml]      = acc0[r] + bv;
      dst[(m0 + 16 + r + hi * 8) * 2 + ml] = acc1[r] + bv;
    }
  }
}

// Issue staging of net-stage s into LDS buffer s&1 (no completion wait here).
// Fast path: 17 uniform async b128 copies per thread (ASYNCcnt).
// Slow path: synchronous f32 -> cvt -> transposed bf16 LDS stores.
__device__ __forceinline__ void stage_issue(int tid, int s, const __bf16* wimg,
    const float* sW1, const float* sb1, const float* sW2, const float* sb2,
    const float* sW3, const float* sb3,
    const float* tW1, const float* tb1, const float* tW2, const float* tb2,
    const float* tW3, const float* tb3,
    __bf16 (*s_w)[IMGSH], float (*s_B1)[HID], float (*s_B2)[HID], float (*s_B3)[2])
{
  const int buf = s & 1;
  const int layer = s >> 1;
  const bool isT = s & 1;
  const float* b1 = (isT ? tb1 : sb1) + layer * 128;
  const float* b2 = (isT ? tb2 : sb2) + layer * 128;
  const float* b3 = (isT ? tb3 : sb3) + layer * 2;
  __bf16* wbuf = s_w[buf];

  if (wimg) {
    const __bf16* img = wimg + (size_t)s * IMGSH;
    const unsigned ldsbase = (unsigned)(uintptr_t)wbuf;   // low 32 bits = LDS offset
    for (int off = tid * 16; off < IMGB; off += 256 * 16) {
      asm volatile("global_load_async_to_lds_b128 %0, %1, %2"
                   :: "v"(ldsbase + (unsigned)off), "v"(off), "s"(img) : "memory");
    }
  } else {
    const float* W1 = (isT ? tW1 : sW1) + layer * 66 * 128;
    const float* W2 = (isT ? tW2 : sW2) + layer * 128 * 128;
    const float* W3 = (isT ? tW3 : sW3) + layer * 256;
    for (int idx = tid; idx < 66 * 128; idx += 256) {      // coalesced read
      int k = idx >> 7, n = idx & 127;
      wbuf[n * P1 + k] = (__bf16)W1[idx];
    }
    for (int idx = tid; idx < 128 * 128; idx += 256) {
      int k = idx >> 7, n = idx & 127;
      wbuf[W2OFF + n * P2 + k] = (__bf16)W2[idx];
    }
    for (int idx = tid; idx < 256; idx += 256) {
      int k = idx >> 1, n = idx & 1;
      wbuf[W3OFF + n * P2 + k] = (__bf16)W3[idx];
    }
  }
  if (tid < 128) { s_B1[buf][tid] = b1[tid]; s_B2[buf][tid] = b2[tid]; }
  if (tid < 2)   { s_B3[buf][tid] = b3[tid]; }
}

// Pre-pass: build 8 net-stage weight images (bf16, transposed, padded) in d_ws.
__global__ void __launch_bounds__(256)
build_weight_images(const float* __restrict__ sW1, const float* __restrict__ sW2,
                    const float* __restrict__ sW3,
                    const float* __restrict__ tW1, const float* __restrict__ tW2,
                    const float* __restrict__ tW3,
                    __bf16* __restrict__ img)
{
  const int s = blockIdx.x;             // 0..7 : stage = layer*2 + (0=s-net,1=t-net)
  const int layer = s >> 1;
  const bool isT = s & 1;
  const float* W1 = (isT ? tW1 : sW1) + layer * 66 * 128;
  const float* W2 = (isT ? tW2 : sW2) + layer * 128 * 128;
  const float* W3 = (isT ? tW3 : sW3) + layer * 256;
  __bf16* o = img + (size_t)s * IMGSH;

  for (int idx = threadIdx.x; idx < IMGSH; idx += 256) {
    float v = 0.0f;
    if (idx < W2OFF) {
      int n = idx / P1, k = idx - n * P1;
      if (k < 66) v = W1[k * 128 + n];
    } else if (idx < W3OFF) {
      int j = idx - W2OFF;
      int n = j / P2, k = j - n * P2;
      if (k < 128) v = W2[k * 128 + n];
    } else {
      int j = idx - W3OFF;
      int n = j / P2, k = j - n * P2;
      if (n < 2 && k < 128) v = W3[k * 2 + n];  // n>=16 (stride pad) stays 0
    }
    o[idx] = (__bf16)v;
  }
}

__global__ void __launch_bounds__(256)
realnvp_logprob_wmma(const float* __restrict__ theta, const float* __restrict__ hctx,
                     const float* sW1, const float* sb1, const float* sW2,
                     const float* sb2, const float* sW3, const float* sb3,
                     const float* tW1, const float* tb1, const float* tW2,
                     const float* tb2, const float* tW3, const float* tb3,
                     float* __restrict__ out, const __bf16* __restrict__ wimg)
{
  __shared__ __align__(16) __bf16 s_xp[BROWS * P1];   // [row][96] padded input
  __shared__ __align__(16) __bf16 s_w[2][IMGSH];      // double-buffered weight image
  __shared__ __align__(16) __bf16 s_h [BROWS * P2];   // hidden activations (reused)
  __shared__ float s_B1[2][HID], s_B2[2][HID], s_B3[2][2];
  __shared__ float s_sv[BROWS * 2], s_tv[BROWS * 2];
  __shared__ float s_x[BROWS][4];

  const int tid  = threadIdx.x;
  const int lane = tid & 31;
  const int m0   = (tid >> 5) * 32;         // wave owns rows [m0, m0+32)
  const long base = (long)blockIdx.x * BROWS;

  // theta -> s_x (f32, persists across layers); one row per thread
  {
    const float4 t4 = ((const float4*)theta)[base + tid];
    s_x[tid][0] = t4.x; s_x[tid][1] = t4.y; s_x[tid][2] = t4.z; s_x[tid][3] = t4.w;
  }
  // h -> xp cols [2,66); coalesced
  for (int idx = tid; idx < BROWS * 64; idx += 256) {
    int r = idx >> 6, c = idx & 63;
    s_xp[r * P1 + 2 + c] = (__bf16)hctx[base * 64 + idx];
  }
  // zero K-pad cols [66,96) of xp (once)
  for (int idx = tid; idx < BROWS * 32; idx += 256) {
    int n = idx >> 5, k = 64 + (idx & 31);
    if (k >= 66) s_xp[n * P1 + k] = (__bf16)0.0f;
  }
  if (!wimg) {  // slow path: pre-zero pads of both weight buffers (never overwritten)
#pragma unroll
    for (int buf = 0; buf < 2; ++buf) {
      for (int idx = tid; idx < HID * 32; idx += 256) {
        int n = idx >> 5, k = 64 + (idx & 31);
        if (k >= 66) s_w[buf][n * P1 + k] = (__bf16)0.0f;
      }
      for (int idx = tid; idx < 14 * 128; idx += 256) {
        int n = 2 + (idx >> 7), k = idx & 127;
        s_w[buf][W3OFF + n * P2 + k] = (__bf16)0.0f;
      }
    }
  }

  const int K0[4] = {0, 1, 2, 0}, K1[4] = {1, 2, 3, 3};
  const int T0[4] = {2, 0, 0, 1}, T1[4] = {3, 3, 1, 2};
  float logdet = 0.0f;

  // xp dynamic cols for layer 0 (keep = {0,1}); row = tid
  s_xp[tid * P1 + 0] = (__bf16)s_x[tid][0];
  s_xp[tid * P1 + 1] = (__bf16)s_x[tid][1];

  // prime stage 0
  stage_issue(tid, 0, wimg, sW1, sb1, sW2, sb2, sW3, sb3,
              tW1, tb1, tW2, tb2, tW3, tb3, s_w, s_B1, s_B2, s_B3);

  for (int s = 0; s < 8; ++s) {
    if (s < 7)      // prefetch next stage into the other buffer (overlaps compute)
      stage_issue(tid, s + 1, wimg, sW1, sb1, sW2, sb2, sW3, sb3,
                  tW1, tb1, tW2, tb2, tW3, tb3, s_w, s_B1, s_B2, s_B3);
    if (wimg) {     // async loads retire in order: <=17 left == stage s complete
      if (s < 7) asm volatile("s_wait_asynccnt 0x11" ::: "memory");
      else       asm volatile("s_wait_asynccnt 0x0"  ::: "memory");
    }
    __syncthreads();                      // stage-s weights/biases visible to all

    const int buf = s & 1;
    __bf16* w1t = s_w[buf];
    __bf16* w2t = s_w[buf] + W2OFF;
    __bf16* w3t = s_w[buf] + W3OFF;
    gemm_tile_act2< 96, P1, P1>(s_xp, w1t, s_B1[buf], s_h, m0, lane);
    gemm_tile_act2<128, P2, P2>(s_h,  w2t, s_B2[buf], s_h, m0, lane);  // in-place, wave-private
    gemm_head2(s_h, w3t, s_B3[buf], (s & 1) ? s_tv : s_sv, m0, lane);
    __syncthreads();                      // outputs visible; buffer free for reuse

    if (s & 1) {                          // coupling update after t-net; row = tid
      const int i = s >> 1;
      float s0 = s_sv[tid * 2 + 0], s1 = s_sv[tid * 2 + 1];
      float t0 = s_tv[tid * 2 + 0], t1 = s_tv[tid * 2 + 1];
      s_x[tid][T0[i]] = s_x[tid][T0[i]] * __expf(s0) + t0;
      s_x[tid][T1[i]] = s_x[tid][T1[i]] * __expf(s1) + t1;
      logdet += s0 + s1;
      if (i < 3) {                        // xp dynamic cols for next layer
        s_xp[tid * P1 + 0] = (__bf16)s_x[tid][K0[i + 1]];
        s_xp[tid * P1 + 1] = (__bf16)s_x[tid][K1[i + 1]];
      }
    }
  }

  __syncthreads();
  {
    float x0 = s_x[tid][0], x1 = s_x[tid][1], x2 = s_x[tid][2], x3 = s_x[tid][3];
    float q = x0 * x0 + x1 * x1 + x2 * x2 + x3 * x3;
    // -0.5*D*log(2*pi) = -2*ln(2*pi)
    out[base + tid] = -0.5f * q - 3.6757541328186907f + logdet;
  }
}

extern "C" void kernel_launch(void* const* d_in, const int* in_sizes, int n_in,
                              void* d_out, int out_size, void* d_ws, size_t ws_size,
                              hipStream_t stream) {
  const float* theta = (const float*)d_in[0];
  const float* hctx  = (const float*)d_in[1];
  const float* sW1 = (const float*)d_in[2];
  const float* sb1 = (const float*)d_in[3];
  const float* sW2 = (const float*)d_in[4];
  const float* sb2 = (const float*)d_in[5];
  const float* sW3 = (const float*)d_in[6];
  const float* sb3 = (const float*)d_in[7];
  const float* tW1 = (const float*)d_in[8];
  const float* tb1 = (const float*)d_in[9];
  const float* tW2 = (const float*)d_in[10];
  const float* tb2 = (const float*)d_in[11];
  const float* tW3 = (const float*)d_in[12];
  const float* tb3 = (const float*)d_in[13];
  float* out = (float*)d_out;

  const int B = in_sizes[0] / 4;          // theta is [B,4]
  const int nblocks = B / BROWS;          // 524288/256 = 2048

  const size_t need = (size_t)8 * IMGSH * sizeof(__bf16);  // 544 KB
  const __bf16* wimg = nullptr;
  if (d_ws && ws_size >= need) {
    build_weight_images<<<dim3(8), dim3(256), 0, stream>>>(
        sW1, sW2, sW3, tW1, tW2, tW3, (__bf16*)d_ws);
    wimg = (const __bf16*)d_ws;
  }

  realnvp_logprob_wmma<<<dim3(nblocks), dim3(256), 0, stream>>>(
      theta, hctx, sW1, sb1, sW2, sb2, sW3, sb3,
      tW1, tb1, tW2, tb2, tW3, tb3, out, wimg);
}
